// CrossAttentionNoGate_48928267436198
// MI455X (gfx1250) — matrix operand order repaired and checked by reading, not compiled, not run on hardware
//
#include <hip/hip_runtime.h>
#include <hip/hip_bf16.h>
#include <math.h>

typedef _Float16 h16;
typedef __attribute__((ext_vector_type(8)))  _Float16 v8h;
typedef __attribute__((ext_vector_type(16))) _Float16 v16h;
typedef __attribute__((ext_vector_type(8)))  float    v8f;

#define S_DIM 128
#define QLEN  256
#define KVLEN 256
#define HEADS 8
#define DH    32
#define CQ    64

// ---- WMMA operand loaders (CDNA5 16x16x32 f16 layouts, wave32) ----

// A tile [16(M) x 32(K)] from f16 row-major buffer, stride ld (halves).
// lanes 0-15: M=lane, K in {0..7, 16..23}; lanes 16-31: M=lane-16, K in {8..15, 24..31}
__device__ __forceinline__ v16h load_a_f16(const h16* base, int ld) {
  const int lane = threadIdx.x & 31;
  const h16* p = base + (lane & 15) * ld + ((lane >> 4) << 3);
  v8h lo = *(const v8h*)p;
  v8h hi = *(const v8h*)(p + 16);
  return __builtin_shufflevector(lo, hi, 0,1,2,3,4,5,6,7,8,9,10,11,12,13,14,15);
}

// Same A tile but converted on the fly from f32 row-major (global inputs).
__device__ __forceinline__ v16h load_a_f32(const float* base, int ld) {
  const int lane = threadIdx.x & 31;
  const float* p = base + (lane & 15) * ld + ((lane >> 4) << 3);
  v16h r;
#pragma unroll
  for (int j = 0; j < 8; ++j) { r[j] = (h16)p[j]; r[8 + j] = (h16)p[16 + j]; }
  return r;
}

// B tile [32(K) x 16(N)] given B^T stored row-major [N x K], stride ld (halves).
// lanes 0-15: N=lane, K in {0..7, 16..23}; lanes 16-31: N=lane-16, K in {8..15, 24..31}
__device__ __forceinline__ v16h load_bt(const h16* base, int ld) {
  const int lane = threadIdx.x & 31;
  const h16* p = base + (lane & 15) * ld + ((lane >> 4) << 3);
  v8h lo = *(const v8h*)p;
  v8h hi = *(const v8h*)(p + 16);
  return __builtin_shufflevector(lo, hi, 0,1,2,3,4,5,6,7,8,9,10,11,12,13,14,15);
}

__device__ __forceinline__ v8f wmma(v16h a, v16h b, v8f c) {
  return __builtin_amdgcn_wmma_f32_16x16x32_f16(false, a, false, b, (short)0, c,
                                                false, false);
}

// ---------------- Kernel 1: fused QKV projection + attention per (s,h) ----------------
__global__ __launch_bounds__(256, 1) void attn_fused(
    const float* __restrict__ xq, const float* __restrict__ xkv,
    const float* __restrict__ mask, const float* __restrict__ bias,
    const float* __restrict__ wq, const float* __restrict__ wk,
    const float* __restrict__ wv, h16* __restrict__ o_ws) {
  __shared__ __align__(16) h16 sm[30720];          // 60 KB
  h16* Kb = sm;                                    // [256][32] K_h row-major
  h16* Vt = sm + 8192;                             // [32][256] V_h transposed
  h16* Qb = sm + 16384;                            // [256][32] Q_h (pre-scaled)
  h16* Wt = sm + 24576;                            // phase A: 3 x [32][64] W^T slices
  h16* Ps = sm + 24576;                            // phase B: per-wave [16][32] P chunk

  const int s    = blockIdx.x >> 3;
  const int h    = blockIdx.x & 7;
  const int tid  = threadIdx.x;
  const int w    = tid >> 5;
  const int lane = tid & 31;
  const int ncol = lane & 15;                      // N index inside a 16-tile
  const int half = lane >> 4;                      // which M/K half this lane owns

  // Load transposed weight slices: Wt[m][n*64+k] = w[k*256 + h*32 + n]
  {
    const float* srcs[3] = {wq, wk, wv};
#pragma unroll
    for (int m = 0; m < 3; ++m) {
      const float* src = srcs[m] + h * 32;
      h16* dst = Wt + m * 2048;
      for (int idx = tid; idx < 2048; idx += 256) {
        int n = idx >> 6, k = idx & 63;
        dst[idx] = (h16)src[k * 256 + n];
      }
    }
  }
  __syncthreads();

  // -------- Phase A: projections. Wave w computes rows [32w, 32w+32). --------
  const float* xq_s  = xq  + (size_t)s * QLEN * CQ;
  const float* xkv_s = xkv + (size_t)s * KVLEN * CQ;
  const float qscale = 0.17677669529663688f;       // 1/sqrt(32)

#pragma unroll
  for (int lm = 0; lm < 2; ++lm) {
    const int m0 = w * 32 + lm * 16;
    v16h aq0 = load_a_f32(xq_s  + m0 * CQ,      CQ);
    v16h aq1 = load_a_f32(xq_s  + m0 * CQ + 32, CQ);
    v16h ak0 = load_a_f32(xkv_s + m0 * CQ,      CQ);
    v16h ak1 = load_a_f32(xkv_s + m0 * CQ + 32, CQ);
#pragma unroll
    for (int nt = 0; nt < 2; ++nt) {
      v8f cq = {}; v8f ck = {}; v8f cv = {};
      cq = wmma(aq0, load_bt(Wt        + nt * 1024,      64), cq);
      cq = wmma(aq1, load_bt(Wt        + nt * 1024 + 32, 64), cq);
      ck = wmma(ak0, load_bt(Wt + 2048 + nt * 1024,      64), ck);
      ck = wmma(ak1, load_bt(Wt + 2048 + nt * 1024 + 32, 64), ck);
      cv = wmma(ak0, load_bt(Wt + 4096 + nt * 1024,      64), cv);
      cv = wmma(ak1, load_bt(Wt + 4096 + nt * 1024 + 32, 64), cv);
#pragma unroll
      for (int i = 0; i < 8; ++i) {
        int m = m0 + half * 8 + i;                 // C-layout: M = 8*half + i
        int n = nt * 16 + ncol;                    //           N = lane%16
        Qb[m * 32 + n]  = (h16)(cq[i] * qscale);
        Kb[m * 32 + n]  = (h16)ck[i];
        Vt[n * 256 + m] = (h16)cv[i];              // store V transposed
      }
    }
  }
  __syncthreads();

  // -------- Phase B: attention. Wave w handles q-tiles 2w, 2w+1. --------
  const float* bias_h = bias + (size_t)h * QLEN * KVLEN;
  const float* mask_s = mask + (size_t)s * KVLEN;
  h16* Pw = Ps + w * 512;                          // per-wave [16][32] scratch

  for (int lm = 0; lm < 2; ++lm) {
    const int q0 = (w * 2 + lm) * 16;
    v16h aq = load_a_f16(Qb + q0 * 32, 32);        // [16 x 32], K = D in one step
    v8f acc[16];
#pragma unroll
    for (int nt = 0; nt < 16; ++nt) {
      // Pre-load C with bias + (mask-1)*INF, then accumulate Q*K^T on top.
      float mv = (mask_s[nt * 16 + ncol] - 1.0f) * 1.0e9f;
#pragma unroll
      for (int i = 0; i < 8; ++i)
        acc[nt][i] = bias_h[(q0 + half * 8 + i) * 256 + nt * 16 + ncol] + mv;
      acc[nt] = wmma(aq, load_bt(Kb + nt * 16 * 32, 32), acc[nt]);
    }

    // Softmax per q-row: row elements live across 16 nt-tiles x 16 lanes of a half.
#pragma unroll
    for (int i = 0; i < 8; ++i) {
      float mx = -3.0e38f;
#pragma unroll
      for (int nt = 0; nt < 16; ++nt) mx = fmaxf(mx, acc[nt][i]);
#pragma unroll
      for (int off = 1; off < 16; off <<= 1) mx = fmaxf(mx, __shfl_xor(mx, off, 32));
      float sum = 0.f;
#pragma unroll
      for (int nt = 0; nt < 16; ++nt) {
        float e = __expf(acc[nt][i] - mx);
        acc[nt][i] = e; sum += e;
      }
#pragma unroll
      for (int off = 1; off < 16; off <<= 1) sum += __shfl_xor(sum, off, 32);
      float r = 1.0f / sum;
#pragma unroll
      for (int nt = 0; nt < 16; ++nt) acc[nt][i] = acc[nt][i] * r;
    }

    // O = P @ V: relayout P (C-layout -> A-layout) through 1KB per-wave LDS chunks.
    v8f o0 = {}; v8f o1 = {};
#pragma unroll
    for (int ks = 0; ks < 8; ++ks) {
#pragma unroll
      for (int c = 0; c < 2; ++c) {
#pragma unroll
        for (int i = 0; i < 8; ++i)
          Pw[(half * 8 + i) * 32 + c * 16 + ncol] = (h16)acc[ks * 2 + c][i];
      }
      v16h ap = load_a_f16(Pw, 32);
      o0 = wmma(ap, load_bt(Vt +            ks * 32, 256), o0);
      o1 = wmma(ap, load_bt(Vt + 16 * 256 + ks * 32, 256), o1);
    }

    // Store O tile (f16) to workspace: row = s*256+q, col = h*32+d
    h16* orow = o_ws + ((size_t)(s * 256 + q0)) * 256 + h * 32;
#pragma unroll
    for (int i = 0; i < 8; ++i) {
      int m = half * 8 + i;
      orow[m * 256 + ncol]      = (h16)o0[i];
      orow[m * 256 + 16 + ncol] = (h16)o1[i];
    }
  }
}

// ---------------- Kernel 2: out = O @ wo + bo ----------------
__global__ __launch_bounds__(256, 1) void out_proj(
    const h16* __restrict__ o_ws, const float* __restrict__ wo,
    const float* __restrict__ bo, float* __restrict__ out) {
  __shared__ __align__(16) h16 wot[64 * 256];      // wo^T: [n][k]
  const int tid = threadIdx.x;
  for (int idx = tid; idx < 64 * 256; idx += 256) {
    int n = idx >> 8, k = idx & 255;
    wot[idx] = (h16)wo[k * 64 + n];
  }
  __syncthreads();

  const int w = tid >> 5, lane = tid & 31;
  const int ncol = lane & 15, half = lane >> 4;
  const int row0 = blockIdx.x * 128 + w * 16;

  v8f acc[4];
#pragma unroll
  for (int nt = 0; nt < 4; ++nt) { v8f z = {}; acc[nt] = z; }

#pragma unroll
  for (int ks = 0; ks < 8; ++ks) {
    v16h a = load_a_f16(o_ws + (size_t)row0 * 256 + ks * 32, 256);
#pragma unroll
    for (int nt = 0; nt < 4; ++nt)
      acc[nt] = wmma(a, load_bt(wot + nt * 16 * 256 + ks * 32, 256), acc[nt]);
  }

#pragma unroll
  for (int nt = 0; nt < 4; ++nt) {
    int col = nt * 16 + ncol;
    float b = bo[col];
#pragma unroll
    for (int i = 0; i < 8; ++i)
      out[(size_t)(row0 + half * 8 + i) * 64 + col] = acc[nt][i] + b;
  }
}

extern "C" void kernel_launch(void* const* d_in, const int* in_sizes, int n_in,
                              void* d_out, int out_size, void* d_ws, size_t ws_size,
                              hipStream_t stream) {
  (void)in_sizes; (void)n_in; (void)out_size; (void)ws_size;
  const float* input_q  = (const float*)d_in[0];
  const float* input_kv = (const float*)d_in[1];
  const float* mask     = (const float*)d_in[2];
  const float* bias     = (const float*)d_in[3];
  const float* wq       = (const float*)d_in[4];
  const float* wk       = (const float*)d_in[5];
  const float* wv       = (const float*)d_in[6];
  const float* wo       = (const float*)d_in[7];
  const float* bo       = (const float*)d_in[8];
  float* out = (float*)d_out;
  h16* o_ws  = (h16*)d_ws;   // [S*Q, H*D] f16 = 16 MiB scratch

  attn_fused<<<dim3(S_DIM * HEADS), dim3(256), 0, stream>>>(
      input_q, input_kv, mask, bias, wq, wk, wv, o_ws);
  out_proj<<<dim3((S_DIM * QLEN) / 128), dim3(256), 0, stream>>>(o_ws, wo, bo, out);
}
